// VARModel_26448408608892
// MI455X (gfx1250) — matrix operand order repaired and checked by reading, not compiled, not run on hardware
//
#include <hip/hip_runtime.h>
#include <hip/hip_bf16.h>
#include <cstdint>

// ---------------------------------------------------------------------------
// CDNA5 (gfx1250) implementation of the VAR transformer reference.
//  * bf16 WMMA (v_wmma_f32_16x16x32_bf16) for all GEMMs + flash attention
//  * global_load_async_to_lds_b128 + s_wait_asynccnt double-buffered staging
//  * weights transposed+converted to bf16 (k-contiguous) once per GEMM
// ---------------------------------------------------------------------------

typedef __bf16 bf16;
typedef bf16  v16bf __attribute__((ext_vector_type(16)));
typedef float v8f   __attribute__((ext_vector_type(8)));

union Frag {
    v16bf    v;
    unsigned u[8];
};

// ---- gfx1250 async copy helpers (inline asm; ISA 15.18.3 opcodes 98/…) ----
__device__ __forceinline__ void async_b128(unsigned lds_byte_off, const void* gaddr)
{
    asm volatile("global_load_async_to_lds_b128 %0, %1, off"
                 :
                 : "v"(lds_byte_off), "v"((unsigned long long)(uintptr_t)gaddr)
                 : "memory");
}
__device__ __forceinline__ void wait_async()
{
    asm volatile("s_wait_asynccnt 0x0" ::: "memory");
}
__device__ __forceinline__ unsigned lds_off(const void* p)
{
    // flat address of an LDS object: addr[31:0] is the LDS byte offset
    return (unsigned)(uintptr_t)p;
}

// ===========================================================================
// Tiled bf16 GEMM:  Out[M,N] = act(A[M,K] @ W[K,N] + bias) (+ res)
//   A  : bf16, row-major, leading dim lda (k-contiguous)
//   Wt : bf16, TRANSPOSED weights Wt[n][k], leading dim ldwt == K
//   Workgroup 256 thr (8 waves); tile 128x128; K-step 32; double-buffered
//   async-to-LDS staging. Wave grid 4(M) x 2(N): 2 Mtiles x 4 Ntiles each.
// flags: bit0 = exact GELU, bit1 = head-mode output permutation,
//        bit2 = bf16 output (else f32)
// ===========================================================================
#define GBM 128
#define GBN 128
#define GBK 32
#define LDSA 40   // padded LDS row (bf16): 80B = 5*16B (async-B128 friendly)
#define LDSB 40

__global__ __launch_bounds__(256)
void k_gemm(const bf16* __restrict__ A, int lda,
            const bf16* __restrict__ Wt, int ldwt,
            const float* __restrict__ bias,
            const float* __restrict__ res,
            void* __restrict__ OutP, int ldo,
            int K, int flags)
{
    __shared__ __align__(16) bf16 sA[2][GBM * LDSA];
    __shared__ __align__(16) bf16 sB[2][GBN * LDSB];

    const int tid = threadIdx.x;
    const int wv  = tid >> 5;
    const int ln  = tid & 31;
    const int wm  = wv & 3;      // 0..3  -> 32 rows each
    const int wn  = wv >> 2;     // 0..1  -> 64 cols each
    const int grp = ln >> 4;     // 0/1
    const int lr  = ln & 15;

    const int m0 = blockIdx.y * GBM;
    const int n0 = blockIdx.x * GBN;

    v8f acc[2][4];
    for (int mt = 0; mt < 2; ++mt)
        for (int nt = 0; nt < 4; ++nt)
            for (int i = 0; i < 8; ++i) acc[mt][nt][i] = 0.f;

    // async stage of one 128x32 A tile + 128x32 Wt tile (4 B128 per thread)
    auto stage = [&](int k0, int buf) {
        #pragma unroll
        for (int i = 0; i < 2; ++i) {
            const int idx  = tid * 2 + i;     // 0..511 B128 slots
            const int row  = idx >> 2;
            const int quad = (idx & 3) << 3;  // bf16 elems
            async_b128(lds_off(&sA[buf][row * LDSA + quad]),
                       A + (size_t)(m0 + row) * lda + k0 + quad);
            async_b128(lds_off(&sB[buf][row * LDSB + quad]),
                       Wt + (size_t)(n0 + row) * ldwt + k0 + quad);
        }
    };

    stage(0, 0);
    wait_async();
    __syncthreads();

    const int KT = K >> 5;
    for (int kt = 0; kt < KT; ++kt) {
        const int cur = kt & 1;
        if (kt + 1 < KT) stage((kt + 1) << 5, cur ^ 1);

        Frag a[2], b[4];
        #pragma unroll
        for (int mt = 0; mt < 2; ++mt) {
            const int row = wm * 32 + mt * 16 + lr;
            #pragma unroll
            for (int v = 0; v < 8; ++v) {
                const int kk = ((v >> 2) << 4) + (grp << 3) + ((v & 3) << 1);
                a[mt].u[v] = *(const unsigned*)&sA[cur][row * LDSA + kk];
            }
        }
        #pragma unroll
        for (int nt = 0; nt < 4; ++nt) {
            const int col = wn * 64 + nt * 16 + lr;
            #pragma unroll
            for (int v = 0; v < 8; ++v) {
                const int kk = ((v >> 2) << 4) + (grp << 3) + ((v & 3) << 1);
                b[nt].u[v] = *(const unsigned*)&sB[cur][col * LDSB + kk];
            }
        }
        #pragma unroll
        for (int mt = 0; mt < 2; ++mt)
            #pragma unroll
            for (int nt = 0; nt < 4; ++nt)
                acc[mt][nt] = __builtin_amdgcn_wmma_f32_16x16x32_bf16(
                    false, a[mt].v, false, b[nt].v, (short)0, acc[mt][nt],
                    false, false);

        wait_async();
        __syncthreads();
    }

    float* Outf = (float*)OutP;
    bf16*  Outh = (bf16*)OutP;
    #pragma unroll
    for (int mt = 0; mt < 2; ++mt)
        #pragma unroll
        for (int nt = 0; nt < 4; ++nt)
            #pragma unroll
            for (int r = 0; r < 8; ++r) {
                const int row = m0 + wm * 32 + mt * 16 + grp * 8 + r;
                const int col = n0 + wn * 64 + nt * 16 + lr;
                float val = acc[mt][nt][r];
                if (bias) val += bias[col];
                if (flags & 1)  // exact GELU
                    val = 0.5f * val * (1.f + erff(val * 0.70710678118f));
                if (res) val += res[(size_t)row * ldo + col];
                if (flags & 2) {
                    // row=(b,n) col=(vocab,c) -> out[b][c][vocab][n]  (f32)
                    const int bb = row >> 10, nn = row & 1023;
                    const int vc = col >> 4,  cc = col & 15;
                    Outf[(((size_t)bb * 16 + cc) * 1024 + vc) * 1024 + nn] = val;
                } else if (flags & 4) {
                    Outh[(size_t)row * ldo + col] = (bf16)val;
                } else {
                    Outf[(size_t)row * ldo + col] = val;
                }
            }
}

// ===========================================================================
// Weight prep: W[K][N] f32 -> Wt[N][K] bf16 (LDS-tiled 32x32 transpose)
// grid (N/32, K/32), block 256 (32x8)
// ===========================================================================
__global__ __launch_bounds__(256)
void k_wt(const float* __restrict__ W, bf16* __restrict__ Wt, int K, int N)
{
    __shared__ float t[32][33];
    const int n0 = blockIdx.x * 32, k0 = blockIdx.y * 32;
    const int tx = threadIdx.x & 31, ty = threadIdx.x >> 5;
    #pragma unroll
    for (int i = 0; i < 4; ++i)
        t[ty + 8 * i][tx] = W[(size_t)(k0 + ty + 8 * i) * N + n0 + tx];
    __syncthreads();
    #pragma unroll
    for (int i = 0; i < 4; ++i)
        Wt[(size_t)(n0 + ty + 8 * i) * K + k0 + tx] = (bf16)t[tx][ty + 8 * i];
}

// ===========================================================================
// LayerNorm over H=1024 per token -> bf16. One 256-thread block per token.
// ===========================================================================
__global__ __launch_bounds__(256)
void k_ln(const float* __restrict__ X, const float* __restrict__ gam,
          const float* __restrict__ bet, bf16* __restrict__ Y)
{
    __shared__ float red[8];
    const size_t t = blockIdx.x;
    const float* xp = X + t * 1024;
    float v[4]; float s = 0.f;
    #pragma unroll
    for (int i = 0; i < 4; ++i) { v[i] = xp[threadIdx.x + 256 * i]; s += v[i]; }
    for (int off = 16; off >= 1; off >>= 1) s += __shfl_xor(s, off, 32);
    const int wv = threadIdx.x >> 5, ln = threadIdx.x & 31;
    if (ln == 0) red[wv] = s;
    __syncthreads();
    s = 0.f;
    #pragma unroll
    for (int i = 0; i < 8; ++i) s += red[i];
    const float mu = s * (1.f / 1024.f);
    float q = 0.f;
    #pragma unroll
    for (int i = 0; i < 4; ++i) { float d = v[i] - mu; q += d * d; }
    for (int off = 16; off >= 1; off >>= 1) q += __shfl_xor(q, off, 32);
    __syncthreads();
    if (ln == 0) red[wv] = q;
    __syncthreads();
    q = 0.f;
    #pragma unroll
    for (int i = 0; i < 8; ++i) q += red[i];
    const float inv = rsqrtf(q * (1.f / 1024.f) + 1e-5f);
    bf16* yp = Y + t * 1024;
    #pragma unroll
    for (int i = 0; i < 4; ++i) {
        const int j = threadIdx.x + 256 * i;
        yp[j] = (bf16)((v[i] - mu) * inv * gam[j] + bet[j]);
    }
}

// ===========================================================================
// f32 -> bf16 convert ([B,N,H] residual stream for the output heads)
// ===========================================================================
__global__ __launch_bounds__(256)
void k_cvt(const float* __restrict__ X, bf16* __restrict__ Y)
{
    const size_t base = (size_t)blockIdx.x * 1024;
    #pragma unroll
    for (int i = 0; i < 4; ++i) {
        const int j = threadIdx.x + 256 * i;
        Y[base + j] = (bf16)X[base + j];
    }
}

// ===========================================================================
// Initial projection + shifted start token (f32 residual stream):
//   h[b,0,:] = cond[b]*Ws + bs ; h[b,n,:] = x[b,n-1,:] @ Wi + bi
// ===========================================================================
__global__ __launch_bounds__(256)
void k_initial(const float* __restrict__ x, const float* __restrict__ cond,
               const float* __restrict__ Wi, const float* __restrict__ bi,
               const float* __restrict__ Ws, const float* __restrict__ bs,
               float* __restrict__ h)
{
    __shared__ float xr[32];
    const int tok = blockIdx.x;
    const int b = tok >> 10, n = tok & 1023;
    if (threadIdx.x < 32)
        xr[threadIdx.x] = (n == 0) ? 0.f
            : x[((size_t)b * 1024 + (n - 1)) * 32 + threadIdx.x];
    __syncthreads();
    #pragma unroll
    for (int i = 0; i < 4; ++i) {
        const int j = threadIdx.x + 256 * i;
        float o;
        if (n == 0) {
            o = cond[b] * Ws[j] + bs[j];
        } else {
            o = bi[j];
            #pragma unroll
            for (int k = 0; k < 32; ++k) o += xr[k] * Wi[k * 1024 + j];
        }
        h[(size_t)tok * 1024 + j] = o;
    }
}

// ===========================================================================
// Split qkv (f32), additive RoPE, emit bf16 [B, nh, N, dh]. Block = 128.
// ===========================================================================
__global__ __launch_bounds__(128)
void k_rope(const float* __restrict__ qkv,
            bf16* __restrict__ q, bf16* __restrict__ k, bf16* __restrict__ v)
{
    const int g = blockIdx.x;                 // (b*8 + h)*1024 + n
    const int n = g & 1023;
    const int hh = (g >> 10) & 7;
    const int b = g >> 13;
    const int d = threadIdx.x;
    const size_t base = ((size_t)(b * 1024 + n)) * 3072 + hh * 128;
    const float tq = qkv[base + 0 * 1024 + d];
    const float tk = qkv[base + 1 * 1024 + d];
    const float tv = qkv[base + 2 * 1024 + d];
    const int dd = d & 63;
    const float inv = __expf(-(float)dd * (9.210340371976184f / 64.f));
    const float emb = (float)n * inv;
    const float sn = __sinf(emb), cs = __cosf(emb);
    const float rq = (d < 64) ? -qkv[base + 0 * 1024 + d + 64]
                              :  qkv[base + 0 * 1024 + d - 64];
    const float rk = (d < 64) ? -qkv[base + 1 * 1024 + d + 64]
                              :  qkv[base + 1 * 1024 + d - 64];
    const size_t ob = (((size_t)b * 8 + hh) * 1024 + n) * 128 + d;
    q[ob] = (bf16)(tq + tq * cs + rq * sn);
    k[ob] = (bf16)(tk + tk * cs + rk * sn);
    v[ob] = (bf16)tv;
}

// ===========================================================================
// Causal flash attention with WMMA (S = Q K^T, O += P V), async Q/K staging.
// Block = 128 (4 waves); 64 query rows per block, key blocks of 32; dh=128.
// Output written bf16 in [B,N,H] layout for the proj GEMM.
// ===========================================================================
#define QLD 136   // 272B row = 17*16B
#define KLD 136
#define VLD 36
#define PLD 40

__global__ __launch_bounds__(128)
void k_flash(const bf16* __restrict__ Q, const bf16* __restrict__ Kg,
             const bf16* __restrict__ Vg, bf16* __restrict__ O)
{
    __shared__ __align__(16) bf16 sQ[64 * QLD];
    __shared__ __align__(16) bf16 sK[32 * KLD];
    __shared__ __align__(16) bf16 sV[128 * VLD];
    __shared__ __align__(16) bf16 sP[4][16 * PLD];

    const int gid  = blockIdx.x;            // (b*8 + h)*16 + qblk
    const int qblk = gid & 15;
    const int bh   = gid >> 4;
    const bf16* qp = Q  + ((size_t)bh * 1024 + qblk * 64) * 128;
    const bf16* kp = Kg + (size_t)bh * 1024 * 128;
    const bf16* vp = Vg + (size_t)bh * 1024 * 128;

    const int tid = threadIdx.x, wv = tid >> 5, ln = tid & 31;
    const int grp = ln >> 4, lr = ln & 15;

    // async stage Q tile 64x128 (1024 B128 slots, 8 per thread)
    #pragma unroll
    for (int i = 0; i < 8; ++i) {
        const int idx  = tid + 128 * i;
        const int row  = idx >> 4;
        const int quad = (idx & 15) << 3;
        async_b128(lds_off(&sQ[row * QLD + quad]), qp + row * 128 + quad);
    }
    wait_async();
    __syncthreads();

    Frag qf[4];
    const int qrow_l = wv * 16 + lr;
    #pragma unroll
    for (int kb = 0; kb < 4; ++kb)
        #pragma unroll
        for (int v = 0; v < 8; ++v) {
            const int kk = kb * 32 + ((v >> 2) << 4) + (grp << 3) + ((v & 3) << 1);
            qf[kb].u[v] = *(const unsigned*)&sQ[qrow_l * QLD + kk];
        }

    v8f o[8];
    float mrow[8], lrow[8];
    #pragma unroll
    for (int nt = 0; nt < 8; ++nt)
        for (int r = 0; r < 8; ++r) o[nt][r] = 0.f;
    #pragma unroll
    for (int r = 0; r < 8; ++r) { mrow[r] = -3.0e38f; lrow[r] = 0.f; }

    const int q0 = qblk * 64;
    const int jmax = (q0 + 63) >> 5;
    const float scale = 0.08838834764831845f;   // 1/sqrt(128)

    for (int j = 0; j <= jmax; ++j) {
        __syncthreads();
        // async K block [key][d] (512 B128 slots, 4 per thread)
        #pragma unroll
        for (int i = 0; i < 4; ++i) {
            const int idx  = tid + 128 * i;
            const int row  = idx >> 4;
            const int quad = (idx & 15) << 3;
            async_b128(lds_off(&sK[row * KLD + quad]),
                       kp + ((size_t)(j * 32 + row)) * 128 + quad);
        }
        // V block transposed: sV[d][key] (VALU path; async cannot transpose)
        #pragma unroll
        for (int i = 0; i < 32; ++i) {
            const int idx = tid * 32 + i;
            const int key = idx >> 7;
            const int dd  = idx & 127;
            sV[dd * VLD + key] = vp[((size_t)(j * 32 + key)) * 128 + dd];
        }
        wait_async();
        __syncthreads();

        // S = Q K^T  (two 16-key N tiles)
        v8f s2[2];
        #pragma unroll
        for (int nt = 0; nt < 2; ++nt)
            for (int r = 0; r < 8; ++r) s2[nt][r] = 0.f;
        #pragma unroll
        for (int nt = 0; nt < 2; ++nt) {
            #pragma unroll
            for (int kb = 0; kb < 4; ++kb) {
                Frag bfr;
                #pragma unroll
                for (int v = 0; v < 8; ++v) {
                    const int kk = kb * 32 + ((v >> 2) << 4) + (grp << 3) + ((v & 3) << 1);
                    bfr.u[v] = *(const unsigned*)&sK[(nt * 16 + lr) * KLD + kk];
                }
                s2[nt] = __builtin_amdgcn_wmma_f32_16x16x32_bf16(
                    false, qf[kb].v, false, bfr.v, (short)0, s2[nt], false, false);
            }
        }

        // streaming softmax
        float fnew[8];
        #pragma unroll
        for (int r = 0; r < 8; ++r) {
            const int row = q0 + wv * 16 + grp * 8 + r;
            float sv0 = s2[0][r] * scale, sv1 = s2[1][r] * scale;
            const int key0 = j * 32 + lr, key1 = j * 32 + 16 + lr;
            if (key0 > row) sv0 = -3.0e38f;
            if (key1 > row) sv1 = -3.0e38f;
            float mc = fmaxf(sv0, sv1);
            for (int off = 1; off < 16; off <<= 1)
                mc = fmaxf(mc, __shfl_xor(mc, off, 32));
            const float mn = fmaxf(mrow[r], mc);
            const float f = __expf(mrow[r] - mn);
            const float p0 = __expf(sv0 - mn), p1 = __expf(sv1 - mn);
            float rs = p0 + p1;
            for (int off = 1; off < 16; off <<= 1)
                rs += __shfl_xor(rs, off, 32);
            lrow[r] = lrow[r] * f + rs;
            mrow[r] = mn;
            fnew[r] = f;
            sP[wv][(grp * 8 + r) * PLD + lr]      = (bf16)p0;
            sP[wv][(grp * 8 + r) * PLD + 16 + lr] = (bf16)p1;
        }
        #pragma unroll
        for (int nt = 0; nt < 8; ++nt)
            for (int r = 0; r < 8; ++r) o[nt][r] *= fnew[r];

        // P as A-fragment (wave-local LDS transpose)
        Frag pf;
        #pragma unroll
        for (int v = 0; v < 8; ++v) {
            const int kk = ((v >> 2) << 4) + (grp << 3) + ((v & 3) << 1);
            pf.u[v] = *(const unsigned*)&sP[wv][lr * PLD + kk];
        }
        // O += P V
        #pragma unroll
        for (int nt = 0; nt < 8; ++nt) {
            Frag bfr;
            #pragma unroll
            for (int v = 0; v < 8; ++v)
                bfr.u[v] = *(const unsigned*)&sV[(nt * 16 + lr) * VLD + ((v >> 2) << 4) + (grp << 3) + ((v & 3) << 1)];
            o[nt] = __builtin_amdgcn_wmma_f32_16x16x32_bf16(
                false, pf.v, false, bfr.v, (short)0, o[nt], false, false);
        }
    }

    // write bf16 atto[b, q, h*128 + d] ([B,N,H])
    const int hh = bh & 7, bb = bh >> 3;
    #pragma unroll
    for (int nt = 0; nt < 8; ++nt)
        #pragma unroll
        for (int r = 0; r < 8; ++r) {
            const int row = q0 + wv * 16 + grp * 8 + r;
            const int dd  = nt * 16 + lr;
            O[((size_t)bb * 1024 + row) * 1024 + hh * 128 + dd] =
                (bf16)(o[nt][r] / lrow[r]);
        }
}

// ===========================================================================
// Orchestration. Input order (insertion order of setup_inputs dicts):
//  0 x | 1 condition | 2-3 initial.w/b | 4-5 start.w/b | 6-7 mag.w/b |
//  8-9 phase.w/b | 8 blocks x {qkv.w,b, proj.w,b, ff1.w,b, ff2.w,b,
//  ln1_g, ln1_b, ln2_g, ln2_b}
// ===========================================================================
extern "C" void kernel_launch(void* const* d_in, const int* in_sizes, int n_in,
                              void* d_out, int out_size, void* d_ws, size_t ws_size,
                              hipStream_t stream)
{
    (void)in_sizes; (void)n_in; (void)out_size; (void)ws_size;
    const float* x    = (const float*)d_in[0];
    const float* cond = (const float*)d_in[1];
    const float* wi   = (const float*)d_in[2];
    const float* bi   = (const float*)d_in[3];
    const float* wst  = (const float*)d_in[4];
    const float* bst  = (const float*)d_in[5];
    const float* wmag = (const float*)d_in[6];
    const float* bmag = (const float*)d_in[7];
    const float* wph  = (const float*)d_in[8];
    const float* bph  = (const float*)d_in[9];

    char* base = (char*)d_ws;
    const size_t MB = 1u << 20;
    float* hA   = (float*)(base + (size_t)  0 * MB);  //  8 MB [B,N,H] f32
    float* hB   = (float*)(base + (size_t)  8 * MB);  //  8 MB
    bf16*  lnb  = (bf16*) (base + (size_t) 16 * MB);  //  4 MB [B,N,H] bf16
    float* qkvb = (float*)(base + (size_t) 20 * MB);  // 24 MB [B,N,3H] f32
    bf16*  atto = (bf16*) (base + (size_t) 44 * MB);  //  4 MB bf16
    bf16*  ffh  = (bf16*) (base + (size_t) 48 * MB);  // 16 MB [B,N,4H] bf16
    bf16*  qb   = (bf16*) (base + (size_t) 64 * MB);  //  4 MB
    bf16*  kb   = (bf16*) (base + (size_t) 68 * MB);  //  4 MB
    bf16*  vb   = (bf16*) (base + (size_t) 72 * MB);  //  4 MB
    bf16*  hbf  = (bf16*) (base + (size_t) 76 * MB);  //  4 MB
    bf16*  wt   = (bf16*) (base + (size_t) 80 * MB);  // 16 MB transposed wts

    k_initial<<<2048, 256, 0, stream>>>(x, cond, wi, bi, wst, bst, hA);

    for (int l = 0; l < 8; ++l) {
        const int p = 10 + l * 12;
        const float* wqkv = (const float*)d_in[p + 0];
        const float* bqkv = (const float*)d_in[p + 1];
        const float* wprj = (const float*)d_in[p + 2];
        const float* bprj = (const float*)d_in[p + 3];
        const float* wff1 = (const float*)d_in[p + 4];
        const float* bff1 = (const float*)d_in[p + 5];
        const float* wff2 = (const float*)d_in[p + 6];
        const float* bff2 = (const float*)d_in[p + 7];
        const float* l1g  = (const float*)d_in[p + 8];
        const float* l1b  = (const float*)d_in[p + 9];
        const float* l2g  = (const float*)d_in[p + 10];
        const float* l2b  = (const float*)d_in[p + 11];

        // qkv
        k_ln<<<2048, 256, 0, stream>>>(hA, l1g, l1b, lnb);
        k_wt<<<dim3(3072 / 32, 1024 / 32), 256, 0, stream>>>(wqkv, wt, 1024, 3072);
        k_gemm<<<dim3(3072 / GBN, 2048 / GBM), 256, 0, stream>>>(
            lnb, 1024, wt, 1024, bqkv, nullptr, qkvb, 3072, 1024, 0);
        // attention
        k_rope<<<2 * 8 * 1024, 128, 0, stream>>>(qkvb, qb, kb, vb);
        k_flash<<<2 * 8 * 16, 128, 0, stream>>>(qb, kb, vb, atto);
        // proj + residual
        k_wt<<<dim3(1024 / 32, 1024 / 32), 256, 0, stream>>>(wprj, wt, 1024, 1024);
        k_gemm<<<dim3(1024 / GBN, 2048 / GBM), 256, 0, stream>>>(
            atto, 1024, wt, 1024, bprj, hA, hB, 1024, 1024, 0);
        // mlp
        k_ln<<<2048, 256, 0, stream>>>(hB, l2g, l2b, lnb);
        k_wt<<<dim3(4096 / 32, 1024 / 32), 256, 0, stream>>>(wff1, wt, 1024, 4096);
        k_gemm<<<dim3(4096 / GBN, 2048 / GBM), 256, 0, stream>>>(
            lnb, 1024, wt, 1024, bff1, nullptr, ffh, 4096, 1024, 1 | 4);
        k_wt<<<dim3(1024 / 32, 4096 / 32), 256, 0, stream>>>(wff2, wt, 4096, 1024);
        k_gemm<<<dim3(1024 / GBN, 2048 / GBM), 256, 0, stream>>>(
            ffh, 4096, wt, 4096, bff2, hB, hA, 1024, 4096, 0);
    }

    // output heads: mag (h cols 0..511), phase (512..1023), head-mode writes
    float* out = (float*)d_out;
    k_cvt<<<2048, 256, 0, stream>>>(hA, hbf);
    k_wt<<<dim3(16384 / 32, 512 / 32), 256, 0, stream>>>(wmag, wt, 512, 16384);
    k_gemm<<<dim3(16384 / GBN, 2048 / GBM), 256, 0, stream>>>(
        hbf, 1024, wt, 512, bmag, nullptr, out, 0, 512, 2);
    k_wt<<<dim3(16384 / 32, 512 / 32), 256, 0, stream>>>(wph, wt, 512, 16384);
    k_gemm<<<dim3(16384 / GBN, 2048 / GBM), 256, 0, stream>>>(
        hbf + 512, 1024, wt, 512, bph, nullptr,
        out + (size_t)2 * 16 * 1024 * 1024, 0, 512, 2);
}
